// DGP_RF_Embeddings_52587579572250
// MI455X (gfx1250) — matrix operand order repaired and checked by reading, not compiled, not run on hardware
//
#include <hip/hip_runtime.h>
#include <hip/hip_bf16.h>

// ---------------- Types ----------------
typedef __attribute__((ext_vector_type(16))) __bf16 bf16x16;
typedef __attribute__((ext_vector_type(8)))  float  f32x8;

union ABPack { bf16x16 v; unsigned int u[8]; };

__device__ __forceinline__ unsigned short f32_to_bf16(float f) {
  unsigned int u = __float_as_uint(f);
  unsigned int r = u + 0x7FFFu + ((u >> 16) & 1u);
  return (unsigned short)(r >> 16);
}

// Async global->LDS copy, 8 bytes per lane (CDNA5, tracked by ASYNCcnt).
__device__ __forceinline__ void async_copy_b64(unsigned lds_byte_off, const void* gaddr) {
  asm volatile("global_load_async_to_lds_b64 %0, %1, off"
               :: "v"(lds_byte_off), "v"(gaddr) : "memory");
}

// ---------------- Constants ----------------
#define NROWS 16384
#define NMC   4
#define RFDIM 512
#define D0    1024
#define D1    512
#define D2    256
#define NATT  4
#define DATT  32
#define BSEG  64

#define LDS_STRIDE 36            // ushorts per 32-elem row (8B-aligned, padded)
#define TILE_USHORTS (64 * LDS_STRIDE)

// ---------------- WMMA GEMM (all-bf16 operands, async double-buffered) ----------------
// C[m] = act(A[m] @ B[m]^T + bias). Block tile 64x64, 8 waves (4M x 2N),
// each wave owns a 16x32 strip = two 16x16 accumulators.
// A: bf16 row-major [M,K]. Bt: bf16 [Ncols,K] (pre-transposed weights).
template <bool HAS_BIAS, bool RELU, bool OUT_BF16>
__global__ __launch_bounds__(256) void gemm_wmma_kernel(
    const unsigned short* __restrict__ Aptr, const unsigned short* __restrict__ Btp,
    const float* __restrict__ bias, void* __restrict__ Cptr,
    unsigned short* __restrict__ Cbf,
    int M, int K, int Ncols,
    long long strideA, long long strideB, long long strideC) {
  __shared__ unsigned short As[2][TILE_USHORTS];
  __shared__ unsigned short Bs[2][TILE_USHORTS];

  const int mb   = blockIdx.z;
  const int row0 = blockIdx.x * 64;
  const int col0 = blockIdx.y * 64;

  const unsigned short* A  = Aptr + mb * strideA;
  const unsigned short* Bt = Btp  + mb * strideB;

  const int tid  = threadIdx.x;
  const int wave = tid >> 5;
  const int lane = tid & 31;
  const int wm   = wave >> 1;   // 0..3 (M)
  const int wn   = wave & 1;    // 0..1 (N)
  const int lm   = lane & 15;
  const int half = lane >> 4;

  // Per-thread staging slots: tile is 64 rows x 8 qwords (32 bf16) per matrix.
  const int ar0 = tid >> 3,        aq0 = tid & 7;          // qword 0..255
  const int ar1 = (tid + 256) >> 3, aq1 = (tid + 256) & 7; // qword 256..511
  const unsigned asbase = (unsigned)(size_t)(&As[0][0]);
  const unsigned bsbase = (unsigned)(size_t)(&Bs[0][0]);

  f32x8 acc[2];
#pragma unroll
  for (int t = 0; t < 2; ++t)
#pragma unroll
    for (int r = 0; r < 8; ++r) acc[t][r] = 0.0f;

  const int T = K >> 5;

  // Prologue: stage tile 0 into buffer 0.
  {
    const unsigned abuf = asbase, bbuf = bsbase;
    async_copy_b64(abuf + (unsigned)(ar0 * LDS_STRIDE + aq0 * 4) * 2,
                   A + (long long)(row0 + ar0) * K + aq0 * 4);
    async_copy_b64(abuf + (unsigned)(ar1 * LDS_STRIDE + aq1 * 4) * 2,
                   A + (long long)(row0 + ar1) * K + aq1 * 4);
    async_copy_b64(bbuf + (unsigned)(ar0 * LDS_STRIDE + aq0 * 4) * 2,
                   Bt + (long long)(col0 + ar0) * K + aq0 * 4);
    async_copy_b64(bbuf + (unsigned)(ar1 * LDS_STRIDE + aq1 * 4) * 2,
                   Bt + (long long)(col0 + ar1) * K + aq1 * 4);
  }

  for (int i = 0; i < T; ++i) {
    const int buf = i & 1;
    if (i + 1 < T) {
      // Stage next tile into the other buffer while this one is consumed.
      const int k1 = (i + 1) << 5;
      const unsigned abuf = asbase + (unsigned)((buf ^ 1) * TILE_USHORTS * 2);
      const unsigned bbuf = bsbase + (unsigned)((buf ^ 1) * TILE_USHORTS * 2);
      async_copy_b64(abuf + (unsigned)(ar0 * LDS_STRIDE + aq0 * 4) * 2,
                     A + (long long)(row0 + ar0) * K + k1 + aq0 * 4);
      async_copy_b64(abuf + (unsigned)(ar1 * LDS_STRIDE + aq1 * 4) * 2,
                     A + (long long)(row0 + ar1) * K + k1 + aq1 * 4);
      async_copy_b64(bbuf + (unsigned)(ar0 * LDS_STRIDE + aq0 * 4) * 2,
                     Bt + (long long)(col0 + ar0) * K + k1 + aq0 * 4);
      async_copy_b64(bbuf + (unsigned)(ar1 * LDS_STRIDE + aq1 * 4) * 2,
                     Bt + (long long)(col0 + ar1) * K + k1 + aq1 * 4);
      asm volatile("s_wait_asynccnt 0x4" ::: "memory");  // tile i has landed
    } else {
      asm volatile("s_wait_asynccnt 0x0" ::: "memory");
    }
    __syncthreads();  // all waves' copies of tile i visible

    // Gather A fragment per ISA 16-bit A 16x32 layout:
    // VGPR g<4: K = 2g + 8*half ; g>=4: K = 16 + 2(g-4) + 8*half
    ABPack a, b0, b1;
    const unsigned short* Asb = &As[buf][0];
    const unsigned short* Bsb = &Bs[buf][0];
    const int arow = (wm * 16 + lm) * LDS_STRIDE;
#pragma unroll
    for (int g = 0; g < 8; ++g) {
      int kk = ((g & 4) << 2) + ((g & 3) << 1) + (half << 3);
      a.u[g] = *(const unsigned int*)(&Asb[arow + kk]);
    }
    // Gather B fragments per ISA B 32x16 layout (same K order as A):
    // VGPR g<4: K = 2g + 8*half ; g>=4: K = 16 + 2(g-4) + 8*half
    const int bc0 = (wn * 32 + lm) * LDS_STRIDE;
    const int bc1 = bc0 + 16 * LDS_STRIDE;
#pragma unroll
    for (int g = 0; g < 8; ++g) {
      int kk = ((g & 4) << 2) + ((g & 3) << 1) + (half << 3);
      b0.u[g] = *(const unsigned int*)(&Bsb[bc0 + kk]);
      b1.u[g] = *(const unsigned int*)(&Bsb[bc1 + kk]);
    }
    acc[0] = __builtin_amdgcn_wmma_f32_16x16x32_bf16(false, a.v, false, b0.v,
                                                     (short)0, acc[0], false, false);
    acc[1] = __builtin_amdgcn_wmma_f32_16x16x32_bf16(false, a.v, false, b1.v,
                                                     (short)0, acc[1], false, false);
    __syncthreads();  // protect this buffer before it is overwritten (i+2)
  }

  // Epilogue: C/D layout -> lane col = lm, VGPR r row = r + 8*half
#pragma unroll
  for (int t = 0; t < 2; ++t) {
    const int col = col0 + wn * 32 + t * 16 + lm;
    float bv = 0.0f;
    if (HAS_BIAS) bv = bias[col];
#pragma unroll
    for (int r = 0; r < 8; ++r) {
      int row = row0 + wm * 16 + r + half * 8;
      float v = acc[t][r] + bv;
      if (RELU) v = v > 0.0f ? v : 0.0f;
      long long ci = mb * strideC + (long long)row * Ncols + col;
      if (OUT_BF16) ((unsigned short*)Cptr)[ci] = f32_to_bf16(v);
      else          ((float*)Cptr)[ci] = v;
      if (Cbf)      Cbf[ci] = f32_to_bf16(v);
    }
  }
}

// ---------------- Operand prep ----------------
__global__ __launch_bounds__(256) void convert_bf16_kernel(
    const float* __restrict__ src, unsigned short* __restrict__ dst, long long n) {
  long long i = (long long)blockIdx.x * 256 + threadIdx.x;
  if (i < n) dst[i] = f32_to_bf16(src[i]);
}

// src fp32 [batch][R][C] -> dst bf16 [batch][C][R] (weights: K-major for GEMM B)
__global__ __launch_bounds__(256) void transpose_bf16_kernel(
    const float* __restrict__ src, unsigned short* __restrict__ dst, int R, int C) {
  __shared__ float tile[32][33];
  const long long base = (long long)blockIdx.z * R * C;
  const int c0 = blockIdx.x * 32, r0 = blockIdx.y * 32;
  const int tx = threadIdx.x & 31, ty = threadIdx.x >> 5;  // 32 x 8
#pragma unroll
  for (int j = 0; j < 32; j += 8)
    tile[ty + j][tx] = src[base + (long long)(r0 + ty + j) * C + c0 + tx];
  __syncthreads();
#pragma unroll
  for (int j = 0; j < 32; j += 8)
    dst[base + (long long)(c0 + ty + j) * R + r0 + tx] = f32_to_bf16(tile[tx][ty + j]);
}

// ---------------- RF features + LayerNorm ----------------
// One block per (m,n) row. z row: 512 fp32 -> phi row: 1024 bf16.
__global__ __launch_bounds__(256) void rf_ln_kernel(const float* __restrict__ z,
                                                    unsigned short* __restrict__ phi) {
  __shared__ float red_s[256];
  __shared__ float red_q[256];
  const long long row = blockIdx.x;
  const int tid = threadIdx.x;
  const float scale = 0.04419417382415922f; // 1/sqrt(512)

  float z0 = z[row * RFDIM + tid];
  float z1 = z[row * RFDIM + tid + 256];
  float s0, c0, s1, c1;
  sincosf(z0, &s0, &c0);
  sincosf(z1, &s1, &c1);
  float x0 = c0 * scale, x1 = s0 * scale, x2 = c1 * scale, x3 = s1 * scale;

  red_s[tid] = x0 + x1 + x2 + x3;
  red_q[tid] = x0 * x0 + x1 * x1 + x2 * x2 + x3 * x3;
  __syncthreads();
#pragma unroll
  for (int off = 128; off > 0; off >>= 1) {
    if (tid < off) {
      red_s[tid] += red_s[tid + off];
      red_q[tid] += red_q[tid + off];
    }
    __syncthreads();
  }
  const float mu  = red_s[0] * (1.0f / 1024.0f);
  const float var = red_q[0] * (1.0f / 1024.0f) - mu * mu;
  const float inv = rsqrtf(var + 1e-5f);

  unsigned short* p = phi + row * 1024;
  p[tid]       = f32_to_bf16((x0 - mu) * inv);
  p[tid + 512] = f32_to_bf16((x1 - mu) * inv);
  p[tid + 256] = f32_to_bf16((x2 - mu) * inv);
  p[tid + 768] = f32_to_bf16((x3 - mu) * inv);
}

// ---------------- Scores: exp((emb@Ws + bs)/16), segment exp-sums ----------------
__global__ __launch_bounds__(256) void scores_kernel(
    const float* __restrict__ emb, const float* __restrict__ Ws,
    const float* __restrict__ bs, const int* __restrict__ X_idx,
    float* __restrict__ escore, float* __restrict__ segsum) {
  const int wave = threadIdx.x >> 5;
  const int lane = threadIdx.x & 31;
  const int p = blockIdx.x * 8 + wave;     // 0 .. N*NMC-1
  const int n = p / NMC;
  const int m = p % NMC;

  const float* e = emb + ((long long)m * NROWS + n) * D2;
  float a0 = 0.f, a1 = 0.f, a2 = 0.f, a3 = 0.f;
#pragma unroll
  for (int i = 0; i < 8; ++i) {
    int d = i * 32 + lane;
    float ev = e[d];
    const float* w = Ws + d * NATT;
    a0 += ev * w[0]; a1 += ev * w[1]; a2 += ev * w[2]; a3 += ev * w[3];
  }
#pragma unroll
  for (int off = 16; off > 0; off >>= 1) {
    a0 += __shfl_xor(a0, off, 32);
    a1 += __shfl_xor(a1, off, 32);
    a2 += __shfl_xor(a2, off, 32);
    a3 += __shfl_xor(a3, off, 32);
  }
  if (lane == 0) {
    float acc[4] = {a0, a1, a2, a3};
    int b = X_idx[n];
#pragma unroll
    for (int k = 0; k < 4; ++k) {
      float v = (acc[k] + bs[k]) * 0.0625f;  // /sqrt(256)
      float ep = expf(v);
      escore[(long long)(n * NMC + m) * 4 + k] = ep;
      atomicAdd(&segsum[(b * NMC + m) * 4 + k], ep);
    }
  }
}

// ---------------- Segment starts (X_idx is sorted) ----------------
__global__ void segstart_kernel(const int* __restrict__ X_idx, int* __restrict__ start) {
  int b = threadIdx.x;
  if (b > BSEG) return;
  int lo = 0, hi = NROWS;
  while (lo < hi) {
    int mid = (lo + hi) >> 1;
    if (X_idx[mid] < b) lo = mid + 1; else hi = mid;
  }
  start[b] = lo;
}

// ---------------- Weighted segment pooling ----------------
__global__ __launch_bounds__(128) void pool_kernel(
    const float* __restrict__ matt, const float* __restrict__ escore,
    const float* __restrict__ segsum, const int* __restrict__ start,
    float* __restrict__ out) {
  const int b = blockIdx.x;
  const int m = blockIdx.y;
  const int e = threadIdx.x;       // 0..127
  const int k = e >> 5;            // attention head

  const int s0 = start[b], s1 = start[b + 1];
  const float rws = 1.0f / segsum[(b * NMC + m) * 4 + k];
  float acc = 0.0f;
  for (int n = s0; n < s1; ++n) {
    float w = escore[(long long)(n * NMC + m) * 4 + k] * rws;
    acc += matt[((long long)m * NROWS + n) * 128 + e] * w;
  }
  out[(b * NMC + m) * 128 + e] = acc;
}

// ---------------- Misc ----------------
__global__ void zero_kernel(float* __restrict__ p, int n) {
  int i = blockIdx.x * blockDim.x + threadIdx.x;
  if (i < n) p[i] = 0.0f;
}

// ---------------- Launch ----------------
extern "C" void kernel_launch(void* const* d_in, const int* in_sizes, int n_in,
                              void* d_out, int out_size, void* d_ws, size_t ws_size,
                              hipStream_t stream) {
  const float* X      = (const float*)d_in[0];
  const int*   X_idx  = (const int*)d_in[1];
  const float* Omega1 = (const float*)d_in[2];
  const float* Omega2 = (const float*)d_in[3];
  const float* W1     = (const float*)d_in[4];
  const float* b1     = (const float*)d_in[5];
  const float* W2     = (const float*)d_in[6];
  const float* b2     = (const float*)d_in[7];
  const float* Ws     = (const float*)d_in[8];
  const float* bs     = (const float*)d_in[9];
  const float* Wm     = (const float*)d_in[10];
  const float* bm     = (const float*)d_in[11];
  float* out = (float*)d_out;

  // Workspace layout
  char* w = (char*)d_ws;
  size_t off = 0;
  float*          zbuf  = (float*)(w + off);          off += (size_t)NMC * NROWS * RFDIM * 4;   // z1/z2
  unsigned short* phi   = (unsigned short*)(w + off); off += (size_t)NMC * NROWS * 1024 * 2;    // phi1/phi2 bf16
  unsigned short* h1    = (unsigned short*)(w + off); off += (size_t)NMC * NROWS * D1 * 2;      // bf16
  float*          emb   = (float*)(w + off);          off += (size_t)NMC * NROWS * D2 * 4;      // fp32
  unsigned short* embb  = (unsigned short*)(w + off); off += (size_t)NMC * NROWS * D2 * 2;      // bf16 copy
  float*          matt  = (float*)(w + off);          off += (size_t)NMC * NROWS * 128 * 4;
  unsigned short* Xb    = (unsigned short*)(w + off); off += (size_t)NROWS * D0 * 2;
  unsigned short* Om1t  = (unsigned short*)(w + off); off += (size_t)NMC * D0 * RFDIM * 2;
  unsigned short* Om2t  = (unsigned short*)(w + off); off += (size_t)NMC * D1 * RFDIM * 2;
  unsigned short* W1t   = (unsigned short*)(w + off); off += (size_t)1024 * D1 * 2;
  unsigned short* W2t   = (unsigned short*)(w + off); off += (size_t)1024 * D2 * 2;
  unsigned short* Wmt   = (unsigned short*)(w + off); off += (size_t)D2 * 128 * 2;
  float*          escr  = (float*)(w + off);          off += (size_t)NROWS * NMC * 4 * 4;
  float*          ssum  = (float*)(w + off);          off += 4096;
  int*            sstar = (int*)(w + off);            off += 512;

  // --- operand prep (bf16 + weight transposes) ---
  convert_bf16_kernel<<<(NROWS * D0 + 255) / 256, 256, 0, stream>>>(X, Xb, (long long)NROWS * D0);
  transpose_bf16_kernel<<<dim3(RFDIM / 32, D0 / 32, NMC), 256, 0, stream>>>(Omega1, Om1t, D0, RFDIM);
  transpose_bf16_kernel<<<dim3(RFDIM / 32, D1 / 32, NMC), 256, 0, stream>>>(Omega2, Om2t, D1, RFDIM);
  transpose_bf16_kernel<<<dim3(D1 / 32, 1024 / 32, 1), 256, 0, stream>>>(W1, W1t, 1024, D1);
  transpose_bf16_kernel<<<dim3(D2 / 32, 1024 / 32, 1), 256, 0, stream>>>(W2, W2t, 1024, D2);
  transpose_bf16_kernel<<<dim3(128 / 32, D2 / 32, 1), 256, 0, stream>>>(Wm, Wmt, D2, 128);
  zero_kernel<<<4, 256, 0, stream>>>(ssum, BSEG * NMC * NATT);

  // z1 = X @ Omega1[m]                       M=16384 K=1024 N=512
  gemm_wmma_kernel<false, false, false><<<dim3(256, 8, 4), 256, 0, stream>>>(
      Xb, Om1t, nullptr, zbuf, nullptr, NROWS, D0, RFDIM,
      0LL, (long long)D0 * RFDIM, (long long)NROWS * RFDIM);

  rf_ln_kernel<<<NMC * NROWS, 256, 0, stream>>>(zbuf, phi);

  // h1 = phi1 @ W1 + b1 (bf16 out)           K=1024 N=512
  gemm_wmma_kernel<true, false, true><<<dim3(256, 8, 4), 256, 0, stream>>>(
      phi, W1t, b1, h1, nullptr, NROWS, 1024, D1,
      (long long)NROWS * 1024, 0LL, (long long)NROWS * D1);

  // z2 = h1 @ Omega2[m]                      K=512 N=512
  gemm_wmma_kernel<false, false, false><<<dim3(256, 8, 4), 256, 0, stream>>>(
      h1, Om2t, nullptr, zbuf, nullptr, NROWS, D1, RFDIM,
      (long long)NROWS * D1, (long long)D1 * RFDIM, (long long)NROWS * RFDIM);

  rf_ln_kernel<<<NMC * NROWS, 256, 0, stream>>>(zbuf, phi);

  // emb = phi2 @ W2 + b2 (fp32 + bf16 out)   K=1024 N=256
  gemm_wmma_kernel<true, false, false><<<dim3(256, 4, 4), 256, 0, stream>>>(
      phi, W2t, b2, emb, embb, NROWS, 1024, D2,
      (long long)NROWS * 1024, 0LL, (long long)NROWS * D2);

  // matt = relu(emb @ Wm + bm)               K=256 N=128
  gemm_wmma_kernel<true, true, false><<<dim3(256, 2, 4), 256, 0, stream>>>(
      embb, Wmt, bm, matt, nullptr, NROWS, D2, NATT * DATT,
      (long long)NROWS * D2, 0LL, (long long)NROWS * 128);

  segstart_kernel<<<1, 128, 0, stream>>>(X_idx, sstar);

  scores_kernel<<<NROWS * NMC / 8, 256, 0, stream>>>(emb, Ws, bs, X_idx, escr, ssum);

  pool_kernel<<<dim3(BSEG, NMC), 128, 0, stream>>>(matt, escr, ssum, sstar, out);

  (void)in_sizes; (void)n_in; (void)out_size; (void)ws_size;
}
